// MHA_14714557956662
// MI455X (gfx1250) — compile-verified, matrix operand order checked by reading, not run on hardware
//
#include <hip/hip_runtime.h>

typedef __attribute__((ext_vector_type(16))) _Float16 v16h;
typedef __attribute__((ext_vector_type(8)))  _Float16 v8h;
typedef __attribute__((ext_vector_type(8)))  float    v8f;

// ---------------------------------------------------------------------------
// Problem constants
// ---------------------------------------------------------------------------
#define C_B    4
#define C_C    512
#define C_F    2
#define C_T    4096
#define C_H    8
#define C_D    64
#define C_BF   8           // B*F
#define C_TKV  4224        // T + R   (R0=64 front pad, R1=64 tail pad)
#define C_TVAL 4160        // T + R0  (valid kv rows)
#define C_NC   32          // T / U
#define C_U    128
#define C_R0   64

// workspace offsets (in halfs)
#define OFF_WQ 0u
#define OFF_WO 786432u            // 1536*512
#define OFF_XT 1048576u           // + 512*512 ; x_t, later reused as o_ws
#define OFF_Q  17825792u          // + 8*4096*512
#define OFF_K  34603008u          // + 8*8*4096*64
#define OFF_V  51904512u          // + 8*8*4224*64

// ---------------------------------------------------------------------------
// WMMA + async helpers
// ---------------------------------------------------------------------------
__device__ __forceinline__ v8f wmma_f16(v16h a, v16h b, v8f c) {
  return __builtin_amdgcn_wmma_f32_16x16x32_f16(
      /*neg_a=*/false, a, /*neg_b=*/false, b,
      /*c_mod=*/(short)0, c, /*reuse_a=*/false, /*reuse_b=*/false);
}

// Async DMA 16B global -> LDS (ASYNCcnt-tracked; CDNA5 GLOBAL_LOAD_ASYNC_TO_LDS_B128)
__device__ __forceinline__ void async_g2l_b128(_Float16* lds_dst, const _Float16* gsrc) {
  unsigned int loff = (unsigned int)(unsigned long long)lds_dst;  // low 32b = LDS offset
  asm volatile("global_load_async_to_lds_b128 %0, %1, off"
               :: "v"(loff), "v"(gsrc) : "memory");
}
__device__ __forceinline__ void wait_async0() {
  asm volatile("s_wait_asynccnt 0" ::: "memory");
}

// Load a 16x32 f16 fragment (A layout; identical pattern serves B when the
// buffer stores B^T, i.e. [N][K] row-major).  Per CDNA5 ISA:
//   lane L: row = L%16, g = L/16
//   elements 0..7  -> K = g*8 .. g*8+7 ; elements 8..15 -> K = 16+g*8 ..
__device__ __forceinline__ v16h load_frag(const _Float16* p, int ld) {
  const int lane = threadIdx.x & 31;
  const int row  = lane & 15;
  const int g    = (lane >> 4) & 1;
  const v8h lo = *(const v8h*)(p + (size_t)row * ld + g * 8);
  const v8h hi = *(const v8h*)(p + (size_t)row * ld + 16 + g * 8);
  v16h r;
#pragma unroll
  for (int i = 0; i < 8; ++i) { r[i] = lo[i]; r[i + 8] = hi[i]; }
  return r;
}

// ---------------------------------------------------------------------------
// Kernel 1: fp32 -> f16 weight conversion
// ---------------------------------------------------------------------------
__global__ void cvt_weights(const float* __restrict__ wqkv,
                            const float* __restrict__ wo,
                            _Float16* __restrict__ ws) {
  int i = blockIdx.x * 256 + threadIdx.x;
  if (i < 786432) ws[OFF_WQ + i] = (_Float16)wqkv[i];
  if (i < 262144) ws[OFF_WO + i] = (_Float16)wo[i];
}

// ---------------------------------------------------------------------------
// Kernel 2: zero padded K/V buffers (34,603,008 halfs = 4,325,376 uint4)
// ---------------------------------------------------------------------------
__global__ void zero_kv(uint4* __restrict__ p) {
  size_t i = (size_t)blockIdx.x * 256 + threadIdx.x;
  uint4 z; z.x = 0u; z.y = 0u; z.z = 0u; z.w = 0u;
  p[i] = z;
}

// ---------------------------------------------------------------------------
// Kernel 3: transpose+convert x (B,C,F,T) fp32 -> x_t[bf][t][c] f16
// ---------------------------------------------------------------------------
__global__ void xpose(const float* __restrict__ x, _Float16* __restrict__ xt) {
  __shared__ float tile[32][33];
  const int bx = blockIdx.x;          // bf*128 + ttile
  const int bf = bx >> 7, tt = bx & 127;
  const int ct = blockIdx.y;
  const int b = bf >> 1, f = bf & 1;
  const int t0 = tt * 32, c0 = ct * 32;
  const int l = threadIdx.x & 31;
  const int q = threadIdx.x >> 5;     // 0..7
#pragma unroll
  for (int p = 0; p < 4; ++p) {
    int cl = q + p * 8;
    tile[cl][l] = x[(((size_t)(b * C_C + c0 + cl) * C_F + f) * C_T) + t0 + l];
  }
  __syncthreads();
#pragma unroll
  for (int p = 0; p < 4; ++p) {
    int tl = q + p * 8;
    xt[((size_t)(bf * C_T + t0 + tl)) * C_C + c0 + l] = (_Float16)tile[l][tl];
  }
}

// ---------------------------------------------------------------------------
// Kernel 4: QKV projection GEMM.  Block tile M=256 (wave = 32 rows, 2 A
// frags, 8 WMMA per k-step), N=64.  Shared B tile (64t x 32c) staged once
// per k-step via async DMA into double-buffered LDS.
// ---------------------------------------------------------------------------
__global__ void qkv_gemm(const _Float16* __restrict__ wq,
                         const _Float16* __restrict__ xt,
                         const float*    __restrict__ bqkv,
                         _Float16* __restrict__ qw,
                         _Float16* __restrict__ kw,
                         _Float16* __restrict__ vw) {
  __shared__ __align__(16) _Float16 bt[2][64][32];   // 2 x 4KB

  int bid = blockIdx.x;
  const int tt = bid & 63; bid >>= 6;
  const int ot = bid % 6;
  const int bf = bid / 6;
  const int tid = threadIdx.x;
  const int w = tid >> 5, lane = tid & 31;
  const int g = (lane >> 4) & 1, l15 = lane & 15;
  const int m0 = ot * 256 + w * 32;
  const int t0 = tt * 64;
  const _Float16* Ab = wq + (size_t)m0 * C_C;
  const _Float16* Bb = xt + ((size_t)bf * C_T + t0) * C_C;

  // cooperative async stage of one 64x32 B tile: 256 threads x 1 b128
  const int srow = tid >> 2;            // 0..63
  const int sch  = (tid & 3) * 8;       // half offset within row
  auto stage = [&](int kk, int buf) {
    async_g2l_b128(&bt[buf][srow][sch], Bb + (size_t)srow * C_C + kk * 32 + sch);
  };

  v8f zero8 = {0.f, 0.f, 0.f, 0.f, 0.f, 0.f, 0.f, 0.f};
  v8f acc[2][4];
#pragma unroll
  for (int mi = 0; mi < 2; ++mi)
#pragma unroll
    for (int i = 0; i < 4; ++i) acc[mi][i] = zero8;

  stage(0, 0);
  v16h a_cur[2], a_nxt[2];
  a_cur[0] = load_frag(Ab, C_C);
  a_cur[1] = load_frag(Ab + 16 * C_C, C_C);

#pragma unroll 2
  for (int kk = 0; kk < 16; ++kk) {
    const int cur = kk & 1;
    wait_async0();          // my async stores to bt[cur] have landed
    __syncthreads();        // everyone's have -> tile complete; prev reads done
    if (kk < 15) stage(kk + 1, cur ^ 1);
    const int kn = (kk < 15) ? kk + 1 : 15;
    a_nxt[0] = load_frag(Ab + kn * 32, C_C);
    a_nxt[1] = load_frag(Ab + 16 * C_C + kn * 32, C_C);
    v16h bfr[4];
#pragma unroll
    for (int nt = 0; nt < 4; ++nt) bfr[nt] = load_frag(&bt[cur][nt * 16][0], 32);
#pragma unroll
    for (int nt = 0; nt < 4; ++nt) {
      acc[0][nt] = wmma_f16(a_cur[0], bfr[nt], acc[0][nt]);
      acc[1][nt] = wmma_f16(a_cur[1], bfr[nt], acc[1][nt]);
    }
    a_cur[0] = a_nxt[0];
    a_cur[1] = a_nxt[1];
  }

#pragma unroll
  for (int mi = 0; mi < 2; ++mi) {
#pragma unroll
    for (int r = 0; r < 8; ++r) {
      const int o = m0 + mi * 16 + r + 8 * g;
      const float bias = bqkv[o];
      const int s = o / C_C, rem = o % C_C;
      const int h = rem >> 6, d = rem & 63;
#pragma unroll
      for (int nt = 0; nt < 4; ++nt) {
        const int t = t0 + nt * 16 + l15;
        const _Float16 hv = (_Float16)(acc[mi][nt][r] + bias);
        if (s == 0)
          qw[(((size_t)bf * C_H + h) * C_T + t) * C_D + d] = hv;
        else if (s == 1)
          kw[(((size_t)bf * C_H + h) * C_TKV + (t + C_R0)) * C_D + d] = hv;
        else
          vw[(((size_t)bf * C_H + h) * C_D + d) * C_TKV + (t + C_R0)] = hv;
      }
    }
  }
}

// ---------------------------------------------------------------------------
// Kernel 5: banded flash attention.  One block = one (bf,h,chunk); 8 waves,
// each wave owns a 16-row query strip.  K/V 64-key blocks shared by all
// waves -> async DMA into double-buffered LDS; key-block loop fully unrolled.
// ---------------------------------------------------------------------------
__global__ void attn(const _Float16* __restrict__ qw,
                     const _Float16* __restrict__ kw,
                     const _Float16* __restrict__ vw,
                     _Float16* __restrict__ ow) {
  __shared__ __align__(16) _Float16 kt[2][64][64];     // 2 x 8KB  [key][d]
  __shared__ __align__(16) _Float16 vt[2][64][64];     // 2 x 8KB  [d][key]
  __shared__ __align__(16) _Float16 ldsP[8][16][64];   // 16KB

  const int bid = blockIdx.x;
  const int c  = bid & 31;
  const int h  = (bid >> 5) & 7;
  const int bf = bid >> 8;
  const int tid = threadIdx.x;
  const int w = tid >> 5, lane = tid & 31;
  const int g = (lane >> 4) & 1, l15 = lane & 15;

  const _Float16* qb = qw + (((size_t)bf * C_H + h) * C_T + c * C_U + w * 16) * C_D;
  const _Float16* kb = kw + ((size_t)bf * C_H + h) * C_TKV * C_D;
  const _Float16* vb = vw + ((size_t)bf * C_H + h) * (size_t)C_D * C_TKV;

  // cooperative async stage of one 64-key block of K and V (8KB each)
  auto stage = [&](int nb, int buf) {
    const int keybase = c * C_U + nb * 64;
#pragma unroll
    for (int it = 0; it < 2; ++it) {
      const int idx = tid + it * 256;     // 0..511
      const int row = idx >> 3;           // 0..63
      const int ch  = (idx & 7) * 8;      // half offset
      async_g2l_b128(&kt[buf][row][ch],
                     kb + (size_t)(keybase + row) * C_D + ch);
      async_g2l_b128(&vt[buf][row][ch],
                     vb + (size_t)row * C_TKV + keybase + ch);
    }
  };

  v16h qf[2];
  qf[0] = load_frag(qb, C_D);
  qf[1] = load_frag(qb + 32, C_D);

  const float NEG = -__builtin_inff();
  float m_i[8], l_i[8], scl[8], rs[8], mb[8];
  v8f zero8 = {0.f, 0.f, 0.f, 0.f, 0.f, 0.f, 0.f, 0.f};
  v8f accO[4];
#pragma unroll
  for (int r = 0; r < 8; ++r) { m_i[r] = NEG; l_i[r] = 0.f; }
#pragma unroll
  for (int dt = 0; dt < 4; ++dt) accO[dt] = zero8;

  stage(0, 0);

#pragma unroll
  for (int nb = 0; nb < 4; ++nb) {
    const int cur = nb & 1;
    wait_async0();
    __syncthreads();
    if (nb < 3) stage(nb + 1, cur ^ 1);

    const int jb = nb * 64;
    // ---- S = Q K^T : preload all 8 K fragments, then WMMA burst ------
    v16h kf[4][2];
#pragma unroll
    for (int nt = 0; nt < 4; ++nt) {
      kf[nt][0] = load_frag(&kt[cur][nt * 16][0], 64);
      kf[nt][1] = load_frag(&kt[cur][nt * 16][32], 64);
    }
    v8f s[4];
#pragma unroll
    for (int nt = 0; nt < 4; ++nt) {
      v8f t = zero8;
      t = wmma_f16(qf[0], kf[nt][0], t);
      t = wmma_f16(qf[1], kf[nt][1], t);
      s[nt] = t;
    }
    // ---- mask + per-block row max -------------------------------------
#pragma unroll
    for (int r = 0; r < 8; ++r) mb[r] = NEG;
#pragma unroll
    for (int nt = 0; nt < 4; ++nt) {
#pragma unroll
      for (int r = 0; r < 8; ++r) {
        const int i_row = w * 16 + r + 8 * g;
        const int j = jb + nt * 16 + l15;
        const int rel = j - i_row;
        float val = s[nt][r] * 0.125f;                 // 1/sqrt(D)
        val = (rel >= 0 && rel <= 128) ? val : -10000.0f;
        if (c * C_U + j >= C_TVAL) val = NEG;
        s[nt][r] = val;
        mb[r] = fmaxf(mb[r], val);
      }
    }
#pragma unroll
    for (int r = 0; r < 8; ++r) {
      mb[r] = fmaxf(mb[r], __shfl_xor(mb[r], 1, 16));
      mb[r] = fmaxf(mb[r], __shfl_xor(mb[r], 2, 16));
      mb[r] = fmaxf(mb[r], __shfl_xor(mb[r], 4, 16));
      mb[r] = fmaxf(mb[r], __shfl_xor(mb[r], 8, 16));
      const float mnew = fmaxf(m_i[r], mb[r]);
      scl[r] = (m_i[r] > NEG) ? expf(m_i[r] - mnew) : 0.0f;
      m_i[r] = mnew;
      rs[r] = 0.f;
    }
    // ---- p = exp(s - m), stage to LDS in f16 --------------------------
#pragma unroll
    for (int nt = 0; nt < 4; ++nt) {
#pragma unroll
      for (int r = 0; r < 8; ++r) {
        const float p = (m_i[r] > NEG && s[nt][r] > NEG)
                            ? expf(s[nt][r] - m_i[r]) : 0.0f;
        rs[r] += p;
        ldsP[w][r + 8 * g][nt * 16 + l15] = (_Float16)p;
      }
    }
#pragma unroll
    for (int r = 0; r < 8; ++r) {
      rs[r] += __shfl_xor(rs[r], 1, 16);
      rs[r] += __shfl_xor(rs[r], 2, 16);
      rs[r] += __shfl_xor(rs[r], 4, 16);
      rs[r] += __shfl_xor(rs[r], 8, 16);
      l_i[r] = l_i[r] * scl[r] + rs[r];
    }
#pragma unroll
    for (int dt = 0; dt < 4; ++dt)
#pragma unroll
      for (int r = 0; r < 8; ++r) accO[dt][r] *= scl[r];
    // ---- O += P @ V : preload V fragments, then WMMA burst ------------
#pragma unroll
    for (int ks = 0; ks < 2; ++ks) {
      v16h a = load_frag(&ldsP[w][0][0] + ks * 32, 64);
      v16h vf[4];
#pragma unroll
      for (int dt = 0; dt < 4; ++dt)
        vf[dt] = load_frag(&vt[cur][dt * 16][ks * 32], 64);
#pragma unroll
      for (int dt = 0; dt < 4; ++dt)
        accO[dt] = wmma_f16(a, vf[dt], accO[dt]);
    }
  }
  // ---- normalize + write o_ws[bf][t][c] f16 ---------------------------
  float inv[8];
#pragma unroll
  for (int r = 0; r < 8; ++r) inv[r] = (l_i[r] > 0.f) ? 1.0f / l_i[r] : 0.f;
#pragma unroll
  for (int dt = 0; dt < 4; ++dt) {
#pragma unroll
    for (int r = 0; r < 8; ++r) {
      const int t  = c * C_U + w * 16 + r + 8 * g;
      const int cd = h * C_D + dt * 16 + l15;
      ow[((size_t)bf * C_T + t) * C_C + cd] = (_Float16)(accO[dt][r] * inv[r]);
    }
  }
}

// ---------------------------------------------------------------------------
// Kernel 6: output projection GEMM, M=256 block tile, fp32 epilogue + bias.
// ---------------------------------------------------------------------------
__global__ void out_gemm(const _Float16* __restrict__ wo,
                         const _Float16* __restrict__ ow,
                         const float*    __restrict__ bo,
                         float* __restrict__ out) {
  __shared__ __align__(16) _Float16 bt[2][64][32];   // 2 x 4KB

  int bid = blockIdx.x;
  const int tt = bid & 63; bid >>= 6;
  const int ot = bid & 1;
  const int bf = bid >> 1;
  const int tid = threadIdx.x;
  const int w = tid >> 5, lane = tid & 31;
  const int g = (lane >> 4) & 1, l15 = lane & 15;
  const int m0 = ot * 256 + w * 32;
  const int t0 = tt * 64;
  const _Float16* Ab = wo + (size_t)m0 * C_C;
  const _Float16* Bb = ow + ((size_t)bf * C_T + t0) * C_C;

  const int srow = tid >> 2;
  const int sch  = (tid & 3) * 8;
  auto stage = [&](int kk, int buf) {
    async_g2l_b128(&bt[buf][srow][sch], Bb + (size_t)srow * C_C + kk * 32 + sch);
  };

  v8f zero8 = {0.f, 0.f, 0.f, 0.f, 0.f, 0.f, 0.f, 0.f};
  v8f acc[2][4];
#pragma unroll
  for (int mi = 0; mi < 2; ++mi)
#pragma unroll
    for (int i = 0; i < 4; ++i) acc[mi][i] = zero8;

  stage(0, 0);
  v16h a_cur[2], a_nxt[2];
  a_cur[0] = load_frag(Ab, C_C);
  a_cur[1] = load_frag(Ab + 16 * C_C, C_C);

#pragma unroll 2
  for (int kk = 0; kk < 16; ++kk) {
    const int cur = kk & 1;
    wait_async0();
    __syncthreads();
    if (kk < 15) stage(kk + 1, cur ^ 1);
    const int kn = (kk < 15) ? kk + 1 : 15;
    a_nxt[0] = load_frag(Ab + kn * 32, C_C);
    a_nxt[1] = load_frag(Ab + 16 * C_C + kn * 32, C_C);
    v16h bfr[4];
#pragma unroll
    for (int nt = 0; nt < 4; ++nt) bfr[nt] = load_frag(&bt[cur][nt * 16][0], 32);
#pragma unroll
    for (int nt = 0; nt < 4; ++nt) {
      acc[0][nt] = wmma_f16(a_cur[0], bfr[nt], acc[0][nt]);
      acc[1][nt] = wmma_f16(a_cur[1], bfr[nt], acc[1][nt]);
    }
    a_cur[0] = a_nxt[0];
    a_cur[1] = a_nxt[1];
  }

  const int b = bf >> 1, f = bf & 1;
#pragma unroll
  for (int mi = 0; mi < 2; ++mi) {
#pragma unroll
    for (int r = 0; r < 8; ++r) {
      const int o = m0 + mi * 16 + r + 8 * g;
      const float bias = bo[o];
#pragma unroll
      for (int nt = 0; nt < 4; ++nt) {
        const int t = t0 + nt * 16 + l15;
        out[(((size_t)b * C_C + o) * C_F + f) * C_T + t] = acc[mi][nt][r] + bias;
      }
    }
  }
}

// ---------------------------------------------------------------------------
extern "C" void kernel_launch(void* const* d_in, const int* in_sizes, int n_in,
                              void* d_out, int out_size, void* d_ws, size_t ws_size,
                              hipStream_t stream) {
  const float* x    = (const float*)d_in[0];
  const float* wqkv = (const float*)d_in[1];
  const float* bqkv = (const float*)d_in[2];
  const float* wo   = (const float*)d_in[3];
  const float* bo   = (const float*)d_in[4];
  float* out = (float*)d_out;

  _Float16* ws = (_Float16*)d_ws;
  _Float16* WQ = ws + OFF_WQ;
  _Float16* WO = ws + OFF_WO;
  _Float16* XT = ws + OFF_XT;   // x_t, later reused as attention output o_ws
  _Float16* QW = ws + OFF_Q;
  _Float16* KW = ws + OFF_K;
  _Float16* VW = ws + OFF_V;

  cvt_weights<<<3072, 256, 0, stream>>>(wqkv, wo, ws);
  zero_kv<<<16896, 256, 0, stream>>>((uint4*)KW);     // zeros K and V regions
  dim3 tg(C_BF * (C_T / 32), C_C / 32);
  xpose<<<tg, 256, 0, stream>>>(x, XT);
  qkv_gemm<<<C_BF * 6 * 64, 256, 0, stream>>>(WQ, XT, bqkv, QW, KW, VW);
  attn<<<C_BF * C_H * C_NC, 256, 0, stream>>>(QW, KW, VW, XT);
  out_gemm<<<C_BF * 2 * 64, 256, 0, stream>>>(WO, XT, bo, out);
}
